// SphereNet_28587302322455
// MI455X (gfx1250) — matrix operand order repaired and checked by reading, not compiled, not run on hardware
//
#include <hip/hip_runtime.h>
#include <hip/hip_bf16.h>

#define NATOMS 512
#define HDIM   128
#define NLAYER 4
#define NRBF   50
#define NMOLS  16

typedef __attribute__((ext_vector_type(16))) __bf16 v16bf;
typedef __attribute__((ext_vector_type(8)))  __bf16 v8bf;
typedef __attribute__((ext_vector_type(8)))  float  v8f;
typedef __attribute__((ext_vector_type(4)))  unsigned int u32x4;
typedef __attribute__((ext_vector_type(4)))  int          i32x4;
typedef __attribute__((ext_vector_type(8)))  int          i32x8;

#if __has_builtin(__builtin_amdgcn_tensor_load_to_lds) && __has_builtin(__builtin_amdgcn_s_wait_tensorcnt)
#define HAVE_TDM 1
#else
#define HAVE_TDM 0
#endif

// ---------- bf16 helpers ----------
__device__ __forceinline__ unsigned short f2bfbits(float f) {
    union { float f; unsigned u; } a; a.f = f;
    unsigned r = (a.u + 0x7FFFu + ((a.u >> 16) & 1u)) >> 16;   // round-to-nearest-even
    return (unsigned short)r;
}
__device__ __forceinline__ __bf16 f2bf(float f) {
    union { unsigned short s; __bf16 b; } o; o.s = f2bfbits(f); return o.b;
}
__device__ __forceinline__ v8bf ld8bf(const unsigned short* p) {
    union { uint4 u; v8bf b; } t;
    t.u = *reinterpret_cast<const uint4*>(p);   // 16B aligned contiguous bf16 x8
    return t.b;
}
__device__ __forceinline__ v16bf cat8(v8bf x, v8bf y) {
    return __builtin_shufflevector(x, y, 0,1,2,3,4,5,6,7,8,9,10,11,12,13,14,15);
}
__device__ __forceinline__ float fast_rcp(float x) {
#if __has_builtin(__builtin_amdgcn_rcpf)
    return __builtin_amdgcn_rcpf(x);            // v_rcp_f32 (approx, plenty for ML fwd)
#else
    return 1.0f / x;
#endif
}
__device__ __forceinline__ float silu(float v) {
    return v * fast_rcp(1.0f + __expf(-v));     // v * sigmoid(v)
}

// ---------- WMMA fragment builders (CDNA5 wave32 layouts, CDNA5 ISA §7.12.2) ----------
// A 16x32 bf16 from fp32 row-major: lane<16 -> K {0..7,16..23}; lane>=16 -> K {8..15,24..31}; M = lane&15.
__device__ __forceinline__ v16bf load_a_f32(const float* __restrict__ A, int lda,
                                            int m0, int k0, int lane) {
    int m  = m0 + (lane & 15);
    int kb = (lane & 16) ? 8 : 0;
    const float* row = A + (size_t)m * lda + k0;
    v16bf a;
#pragma unroll
    for (int e = 0; e < 8; ++e)  a[e]     = f2bf(row[kb + e]);
#pragma unroll
    for (int e = 0; e < 8; ++e)  a[8 + e] = f2bf(row[16 + kb + e]);
    return a;
}
// B 32x16 bf16 from fp32 row-major KxN (no bounds checks; K multiple of 32 or pre-padded).
// lane<16 -> K {0..7,16..23}, lane>=16 -> K {8..15,24..31}; N = lane&15.
__device__ __forceinline__ v16bf load_b_f32(const float* __restrict__ B, int ldb,
                                            int k0, int n0, int lane) {
    int n  = n0 + (lane & 15);
    const float* col = B + (size_t)(k0 + ((lane & 16) ? 8 : 0)) * ldb + n;
    v16bf b;
#pragma unroll
    for (int v = 0; v < 8; ++v) {
        b[2*v]   = f2bf(col[(size_t)((2*v)     < 8 ? (2*v)     : (2*v) + 8)     * ldb]);
        b[2*v+1] = f2bf(col[(size_t)((2*v + 1) < 8 ? (2*v + 1) : (2*v + 1) + 8) * ldb]);
    }
    return b;
}

// ---------- TDM helpers ----------
#if HAVE_TDM
__device__ __forceinline__ void tdm_call(u32x4 g0, i32x8 g1) {
    i32x4 z4 = {0, 0, 0, 0};
#if defined(__clang_major__) && __clang_major__ >= 23
    i32x8 z8 = {0, 0, 0, 0, 0, 0, 0, 0};
    __builtin_amdgcn_tensor_load_to_lds(g0, g1, z4, z4, z8, 0);
#else
    __builtin_amdgcn_tensor_load_to_lds(g0, g1, z4, z4, 0);
#endif
}
// Load 8 consecutive i-rows of one 16-j tile: 2D tile, X = 1024 bf16 elements (16 j * 64 feat),
// Y = 8 (i), Y-stride = NATOMS*64 elements. Destination: LDS at lds_off bytes.
__device__ __forceinline__ void tdm_issue_chunk(const unsigned short* geo, int j0,
                                                int i0, unsigned lds_off) {
    unsigned long long ga =
        (unsigned long long)(size_t)(geo + ((size_t)i0 * NATOMS + j0) * 64);
    u32x4 g0;
    g0[0] = 1u;                                            // count=1, user mode
    g0[1] = lds_off;                                       // lds_addr (bytes)
    g0[2] = (unsigned)(ga & 0xFFFFFFFFu);                  // global_addr[31:0]
    g0[3] = (unsigned)((ga >> 32) & 0x01FFFFFFu)           // global_addr[56:32]
          | 0x80000000u;                                   // type=2 ("image") bits 127:126
    i32x8 g1;
    g1[0] = 1 << 16;                                       // data_size=1 (2 bytes), no flags
    g1[1] = (int)(1024u << 16);                            // tensor_dim0[15:0]
    g1[2] = (int)(8u << 16);                               // tensor_dim0[31:16]=0 | tensor_dim1[15:0]=8
    g1[3] = (int)(1024u << 16);                            // tensor_dim1[31:16]=0 | tile_dim0=1024
    g1[4] = 8;                                             // tile_dim1=8, tile_dim2=0
    g1[5] = NATOMS * 64;                                   // tensor_dim0_stride[31:0]
    g1[6] = 0;                                             // stride0[47:32] | dim1_stride[15:0]
    g1[7] = 0;                                             // dim1_stride[47:16]
    tdm_call(g0, g1);
}
#endif

// ---------- prep: geometry features (layer-invariant) ----------
__global__ void prep_geo_kernel(const float* __restrict__ pos,
                                unsigned short* __restrict__ geo,     // [N*N, 64] bf16
                                unsigned char* __restrict__ vbyte) {  // [N*N] validity
    int p = blockIdx.x * blockDim.x + threadIdx.x;
    if (p >= NATOMS * NATOMS) return;
    int i = p >> 9, j = p & (NATOMS - 1);
    float dx = pos[i*3+0] - pos[j*3+0];
    float dy = pos[i*3+1] - pos[j*3+1];
    float dz = pos[i*3+2] - pos[j*3+2];
    float dist = sqrtf(dx*dx + dy*dy + dz*dz);
    float inv  = fast_rcp(fmaxf(dist, 1e-8f));
    unsigned short* g = geo + (size_t)p * 64;
    const float width  = 5.0f / (float)NRBF;            // 0.1
    const float i2w2   = 1.0f / (2.0f * width * width);
    const float step   = 5.0f / (float)(NRBF - 1);      // linspace(0,5,50)
#pragma unroll
    for (int k = 0; k < NRBF; ++k) {
        float t = dist - step * (float)k;
        g[k] = f2bfbits(__expf(-t * t * i2w2));
    }
    g[50] = f2bfbits(dx * inv);
    g[51] = f2bfbits(dy * inv);
    g[52] = f2bfbits(dz * inv);
#pragma unroll
    for (int k = 53; k < 64; ++k) g[k] = 0;
    vbyte[p] = (unsigned char)((dist < 5.0f) && (i != j));
}

__global__ void pack_mask_kernel(const unsigned char* __restrict__ vbyte,
                                 unsigned* __restrict__ vmask) {       // [N, 16] words
    int t = blockIdx.x * blockDim.x + threadIdx.x;
    if (t >= NATOMS * 16) return;
    const unsigned char* b = vbyte + (size_t)(t >> 4) * NATOMS + (t & 15) * 32;
    unsigned m = 0;
#pragma unroll
    for (int k = 0; k < 32; ++k) m |= (unsigned)(b[k] & 1) << k;
    vmask[t] = m;
}

__global__ void counts_kernel(const unsigned char* __restrict__ vbyte,
                              float* __restrict__ counts) {            // [N]
    int j = blockIdx.x * blockDim.x + threadIdx.x;
    if (j >= NATOMS) return;
    int c = 0;
    for (int i = 0; i < NATOMS; ++i) c += vbyte[(size_t)i * NATOMS + j];
    counts[j] = (float)c;
}

__global__ void embed_kernel(const int* __restrict__ an,
                             const float* __restrict__ embed,
                             float* __restrict__ x) {
    int t = blockIdx.x * blockDim.x + threadIdx.x;
    if (t >= NATOMS * HDIM) return;
    int n = t >> 7, h = t & (HDIM - 1);
    int a = an[n]; a = a < 0 ? 0 : (a > 99 ? 99 : a);
    x[t] = embed[a * HDIM + h];
}

// zero-padded copy of msg_w1 rows 128..180 -> wrdpad[64][128]
__global__ void pad_wrd_kernel(const float* __restrict__ W1, float* __restrict__ wrdpad) {
    int t = blockIdx.x * blockDim.x + threadIdx.x;
    if (t >= 64 * HDIM) return;
    int r = t >> 7;
    wrdpad[t] = (r < 53) ? W1[(size_t)(HDIM + r) * HDIM + (t & (HDIM - 1))] : 0.0f;
}

// ---------- generic bf16 WMMA GEMM: C = act(A@B + bias(*rowscale) + Cin) ----------
// A: MxK fp32 (row-major), B: KxN fp32 (row-major). grid = (N/128, M/16), block = 256 (8 waves).
// K is a compile-time multiple of 32 -> fully unrolled, unguarded loads.
template <int K>
__global__ void gemm_kernel(const float* __restrict__ A, int lda,
                            const float* __restrict__ B, int ldb,
                            const float* __restrict__ bias,       // per-col, may be null
                            const float* __restrict__ rowscale,   // per-row scale of bias, may be null
                            const float* __restrict__ Cin, int ldcin,  // residual add, may be null
                            float* __restrict__ C, int ldc, int act) {
    int lane = threadIdx.x & 31;
    int wave = threadIdx.x >> 5;
    int n0 = blockIdx.x * 128 + wave * 16;
    int m0 = blockIdx.y * 16;

    v8f acc;
#pragma unroll
    for (int r = 0; r < 8; ++r) acc[r] = 0.0f;

#pragma unroll
    for (int k0 = 0; k0 < K; k0 += 32) {
        v16bf a = load_a_f32(A, lda, m0, k0, lane);
        v16bf b = load_b_f32(B, ldb, k0, n0, lane);
        acc = __builtin_amdgcn_wmma_f32_16x16x32_bf16(false, a, false, b,
                                                      (short)0, acc, false, false);
    }

    int n  = n0 + (lane & 15);
    int mb = m0 + ((lane & 16) ? 8 : 0);
    float bs = bias ? bias[n] : 0.0f;
#pragma unroll
    for (int r = 0; r < 8; ++r) {
        int m = mb + r;
        float v = acc[r] + (rowscale ? rowscale[m] * bs : bs);
        if (Cin) v += Cin[(size_t)m * ldcin + n];
        if (act) v = silu(v);
        C[(size_t)m * ldc + n] = v;
    }
}

// ---------- hot loop: S[j,h] = sum_i valid[i,j] * silu(xW[i,h] + geo[i,j]@Wrd[:,h]) ----------
// grid.x = N/16 (j tiles), block = 256 (8 waves, wave w owns h-tile w*16..w*16+15).
// geo tiles are staged through LDS in chunks of 8 i-rows (16KB), double-buffered:
// TDM (tensor_load_to_lds + s_wait_tensorcnt) when available, cooperative copy otherwise.
#define CHUNK    8
#define TILE_HLF (16 * 64)              // bf16 elements per i-row tile
#define CHUNK_HLF (CHUNK * TILE_HLF)    // 8192 bf16 = 16KB

__global__ void msg_kernel(const unsigned short* __restrict__ geo,  // [N*N,64] bf16
                           const float* __restrict__ wrdpad,        // [64,128] fp32, zero-padded
                           const float* __restrict__ xW,            // [N,128] = x@Wx + b1
                           const unsigned* __restrict__ vmask,      // [N,16]
                           float* __restrict__ S) {                 // [N,128]
    __shared__ __align__(16) unsigned short gbuf[2][CHUNK_HLF];

    int lane = threadIdx.x & 31;
    int wave = threadIdx.x >> 5;
    int h0 = wave * 16;
    int j0 = blockIdx.x * 16;

    // loop-invariant B fragments (Wrd padded to 64 rows)
    v16bf bf0 = load_b_f32(wrdpad, HDIM, 0,  h0, lane);
    v16bf bf1 = load_b_f32(wrdpad, HDIM, 32, h0, lane);

    v8f acc;
#pragma unroll
    for (int r = 0; r < 8; ++r) acc[r] = 0.0f;

    int m     = lane & 15;                  // j_local (A-matrix row)
    int kb    = (lane & 16) ? 8 : 0;        // A-layout K sub-block
    int hn    = h0 + (lane & 15);           // this lane's hidden index (N dim)
    int jbit0 = (j0 & 31) + ((lane & 16) ? 8 : 0);
    const unsigned* mrow = vmask + (j0 >> 5);

#if HAVE_TDM
    unsigned lds0 = (unsigned)(size_t)&gbuf[0][0];
    unsigned lds1 = (unsigned)(size_t)&gbuf[1][0];
    if (wave == 0) tdm_issue_chunk(geo, j0, 0, lds0);
#endif

    for (int c = 0; c < NATOMS / CHUNK; ++c) {
#if HAVE_TDM
        if (wave == 0) {
            if (c + 1 < NATOMS / CHUNK) {
                tdm_issue_chunk(geo, j0, (c + 1) * CHUNK, ((c + 1) & 1) ? lds1 : lds0);
                __builtin_amdgcn_s_wait_tensorcnt(1);   // chunk c has landed (in-order)
            } else {
                __builtin_amdgcn_s_wait_tensorcnt(0);
            }
        }
        __syncthreads();                                // chunk c visible to all waves
#else
        {   // cooperative copy: 256 threads x 32 halfs = 16KB
            const unsigned short* src = geo + ((size_t)(c * CHUNK) * NATOMS + j0) * 64;
            uint4* dst = reinterpret_cast<uint4*>(&gbuf[c & 1][0]);
            const uint4* s4 = reinterpret_cast<const uint4*>(src);
#pragma unroll
            for (int q = 0; q < 4; ++q)
                dst[threadIdx.x + 256 * q] = s4[threadIdx.x + 256 * q];
            if (c + 1 < NATOMS / CHUNK)
                __builtin_prefetch(geo + ((size_t)((c + 1) * CHUNK) * NATOMS + j0) * 64, 0, 1);
            __syncthreads();
        }
#endif
        const unsigned short* lbase = &gbuf[c & 1][0];
#pragma unroll
        for (int ii = 0; ii < CHUNK; ++ii) {
            int i = c * CHUNK + ii;
            const unsigned short* lrow = lbase + ii * TILE_HLF + m * 64;

            v16bf a0 = cat8(ld8bf(lrow + kb),      ld8bf(lrow + 16 + kb));
            v16bf a1 = cat8(ld8bf(lrow + 32 + kb), ld8bf(lrow + 48 + kb));

            v8f p;
#pragma unroll
            for (int r = 0; r < 8; ++r) p[r] = 0.0f;
            p = __builtin_amdgcn_wmma_f32_16x16x32_bf16(false, a0, false, bf0, (short)0, p, false, false);
            p = __builtin_amdgcn_wmma_f32_16x16x32_bf16(false, a1, false, bf1, (short)0, p, false, false);

            float xwv   = xW[(size_t)i * HDIM + hn];   // includes b1
            unsigned mw = mrow[i * 16];
#pragma unroll
            for (int r = 0; r < 8; ++r) {
                float pre = p[r] + xwv;
                float sv  = silu(pre);
                if ((mw >> (jbit0 + r)) & 1u) acc[r] += sv;
            }
        }
        __syncthreads();    // all reads of gbuf[c&1] done before it is refilled
    }

#pragma unroll
    for (int r = 0; r < 8; ++r) {
        int j = j0 + r + ((lane & 16) ? 8 : 0);
        S[(size_t)j * HDIM + hn] = acc[r];
    }
}

// ---------- pooling + output head (tiny) ----------
__global__ void pool_kernel(const float* __restrict__ x, const int* __restrict__ batch,
                            float* __restrict__ pooled) {
    int mol = blockIdx.x, h = threadIdx.x;
    float s = 0.0f; int c = 0;
    for (int n = 0; n < NATOMS; ++n) {
        if (batch[n] == mol) { s += x[(size_t)n * HDIM + h]; ++c; }
    }
    pooled[mol * HDIM + h] = s * fast_rcp(fmaxf((float)c, 1.0f));
}

__global__ void out_kernel(const float* __restrict__ pooled,
                           const float* __restrict__ w1, const float* __restrict__ b1,
                           const float* __restrict__ w2, const float* __restrict__ b2,
                           float* __restrict__ out) {
    __shared__ float h1[NMOLS * 64];
    for (int t = threadIdx.x; t < NMOLS * 64; t += blockDim.x) {
        int mol = t >> 6, u = t & 63;
        float s = b1[u];
        for (int k = 0; k < HDIM; ++k) s += pooled[mol * HDIM + k] * w1[k * 64 + u];
        h1[t] = silu(s);
    }
    __syncthreads();
    for (int t = threadIdx.x; t < NMOLS; t += blockDim.x) {
        float s = b2[0];
        for (int u = 0; u < 64; ++u) s += h1[t * 64 + u] * w2[u];
        out[t] = s;
    }
}

// ---------- host orchestration ----------
extern "C" void kernel_launch(void* const* d_in, const int* in_sizes, int n_in,
                              void* d_out, int out_size, void* d_ws, size_t ws_size,
                              hipStream_t stream) {
    const int*   an     = (const int*)  d_in[0];
    const float* pos    = (const float*)d_in[1];
    const int*   batch  = (const int*)  d_in[2];
    const float* embedw = (const float*)d_in[3];
    const float* msg_w1 = (const float*)d_in[4];   // L x 181 x 128
    const float* msg_b1 = (const float*)d_in[5];
    const float* msg_w2 = (const float*)d_in[6];   // L x 128 x 128
    const float* msg_b2 = (const float*)d_in[7];
    const float* upd_w1 = (const float*)d_in[8];   // L x 256 x 128
    const float* upd_b1 = (const float*)d_in[9];
    const float* upd_w2 = (const float*)d_in[10];  // L x 128 x 128
    const float* upd_b2 = (const float*)d_in[11];
    const float* out_w1 = (const float*)d_in[12];  // 128 x 64
    const float* out_b1 = (const float*)d_in[13];
    const float* out_w2 = (const float*)d_in[14];  // 64 x 1
    const float* out_b2 = (const float*)d_in[15];
    float* out = (float*)d_out;
    (void)in_sizes; (void)n_in; (void)out_size; (void)ws_size;

    char* ws = (char*)d_ws;
    size_t off = 0;
    auto alloc = [&](size_t bytes) -> void* {
        void* p = ws + off;
        off += (bytes + 255) & ~(size_t)255;
        return p;
    };
    unsigned short* geo    = (unsigned short*)alloc((size_t)NATOMS * NATOMS * 64 * 2);
    unsigned char*  vbyte  = (unsigned char*) alloc((size_t)NATOMS * NATOMS);
    unsigned*       vmask  = (unsigned*)      alloc((size_t)NATOMS * 16 * 4);
    float*          counts = (float*)         alloc((size_t)NATOMS * 4);
    float*          wrdpad = (float*)         alloc((size_t)64 * HDIM * 4);
    float*          xA     = (float*)         alloc((size_t)NATOMS * HDIM * 4);
    float*          xB     = (float*)         alloc((size_t)NATOMS * HDIM * 4);
    float*          xW     = (float*)         alloc((size_t)NATOMS * HDIM * 4);
    float*          Sbuf   = (float*)         alloc((size_t)NATOMS * HDIM * 4);
    float*          aggr   = (float*)         alloc((size_t)NATOMS * HDIM * 4);
    float*          tbuf   = (float*)         alloc((size_t)NATOMS * HDIM * 4);
    float*          h1buf  = (float*)         alloc((size_t)NATOMS * HDIM * 4);
    float*          pooled = (float*)         alloc((size_t)NMOLS * HDIM * 4);

    // ---- precompute (layer-invariant) ----
    prep_geo_kernel<<<(NATOMS * NATOMS) / 256, 256, 0, stream>>>(pos, geo, vbyte);
    pack_mask_kernel<<<(NATOMS * 16 + 255) / 256, 256, 0, stream>>>(vbyte, vmask);
    counts_kernel<<<(NATOMS + 255) / 256, 256, 0, stream>>>(vbyte, counts);
    embed_kernel<<<(NATOMS * HDIM) / 256, 256, 0, stream>>>(an, embedw, xA);

    dim3 g128(1, NATOMS / 16);   // GEMM grid for N=128, M=512
    float* x  = xA;
    float* xn = xB;
    for (int l = 0; l < NLAYER; ++l) {
        const float* W1  = msg_w1 + (size_t)l * 181 * HDIM;
        const float* b1  = msg_b1 + (size_t)l * HDIM;
        const float* W2  = msg_w2 + (size_t)l * HDIM * HDIM;
        const float* b2  = msg_b2 + (size_t)l * HDIM;
        const float* U1  = upd_w1 + (size_t)l * 2 * HDIM * HDIM;
        const float* ub1 = upd_b1 + (size_t)l * HDIM;
        const float* U2  = upd_w2 + (size_t)l * HDIM * HDIM;
        const float* ub2 = upd_b2 + (size_t)l * HDIM;

        // zero-padded Wrd for this layer (rows 128..180 of W1)
        pad_wrd_kernel<<<(64 * HDIM) / 256, 256, 0, stream>>>(W1, wrdpad);
        // xW = x @ Wx + b1   (Wx = W1 rows 0..127)
        gemm_kernel<128><<<g128, 256, 0, stream>>>(x, HDIM, W1, HDIM, b1, nullptr,
                                                   nullptr, 0, xW, HDIM, 0);
        // S[j,h] = sum_i valid * silu(pre)   (hot kernel)
        msg_kernel<<<NATOMS / 16, 256, 0, stream>>>(geo, wrdpad, xW, vmask, Sbuf);
        // aggr = S @ W2 + counts[j]*b2
        gemm_kernel<128><<<g128, 256, 0, stream>>>(Sbuf, HDIM, W2, HDIM, b2, counts,
                                                   nullptr, 0, aggr, HDIM, 0);
        // tbuf = x @ U1_top
        gemm_kernel<128><<<g128, 256, 0, stream>>>(x, HDIM, U1, HDIM, nullptr, nullptr,
                                                   nullptr, 0, tbuf, HDIM, 0);
        // h1 = silu(aggr @ U1_bot + tbuf + ub1)
        gemm_kernel<128><<<g128, 256, 0, stream>>>(aggr, HDIM, U1 + (size_t)HDIM * HDIM, HDIM,
                                                   ub1, nullptr, tbuf, HDIM, h1buf, HDIM, 1);
        // xn = x + h1 @ U2 + ub2
        gemm_kernel<128><<<g128, 256, 0, stream>>>(h1buf, HDIM, U2, HDIM, ub2, nullptr,
                                                   x, HDIM, xn, HDIM, 0);
        float* t = x; x = xn; xn = t;
    }

    pool_kernel<<<NMOLS, HDIM, 0, stream>>>(x, batch, pooled);
    out_kernel<<<1, 256, 0, stream>>>(pooled, out_w1, out_b1, out_w2, out_b2, out);
}
